// NARSelfAttention_88742614270524
// MI455X (gfx1250) — matrix operand order repaired and checked by reading, not compiled, not run on hardware
//
#include <hip/hip_runtime.h>

// ---------------------------------------------------------------------------
// GQA self-attention for gfx1250 (MI455X): bf16 WMMA everywhere, async-to-LDS
// tile pipeline in the GEMMs (global_load_async_to_lds_b128, ASYNCcnt).
// B=2, S=2048, D=2048, NH=32, NKV=8, HD=64. qpk = 4, head h uses kv-head h/4.
// ---------------------------------------------------------------------------

typedef __bf16 bf16;
typedef __attribute__((ext_vector_type(16))) __bf16 v16bf;
typedef __attribute__((ext_vector_type(8)))  __bf16 v8bf;
typedef __attribute__((ext_vector_type(8)))  float  v8f;
typedef int v4i_vs __attribute__((vector_size(16)));
typedef __attribute__((address_space(1))) v4i_vs* gptr_b128;
typedef __attribute__((address_space(3))) v4i_vs* lptr_b128;

#define ZERO8 ((v8f){0.f,0.f,0.f,0.f,0.f,0.f,0.f,0.f})

#if defined(__has_builtin)
#if __has_builtin(__builtin_amdgcn_global_load_async_to_lds_b128)
#define HAVE_ASYNC_LDS 1
#endif
#endif
#ifndef HAVE_ASYNC_LDS
#define HAVE_ASYNC_LDS 0
#endif

// 16-byte global -> LDS copy. Async (ASYNCcnt-tracked, bypasses VGPRs) when
// available; otherwise classic load + ds_store (DScnt, flushed by barrier).
__device__ __forceinline__ void copy16_g2l(const bf16* g, bf16* l) {
#if HAVE_ASYNC_LDS
  __builtin_amdgcn_global_load_async_to_lds_b128(
      (gptr_b128)g, (lptr_b128)l, 0, 0);
#else
  *(uint4*)l = *(const uint4*)g;
#endif
}

__device__ __forceinline__ void wait_async_le4() {
#if HAVE_ASYNC_LDS
  asm volatile("s_wait_asynccnt 0x4" ::: "memory");
#endif
}
__device__ __forceinline__ void wait_async_0() {
#if HAVE_ASYNC_LDS
  asm volatile("s_wait_asynccnt 0x0" ::: "memory");
#endif
}

__device__ __forceinline__ v16bf pack16(v8bf lo, v8bf hi) {
  v16bf r;
#pragma unroll
  for (int i = 0; i < 8; ++i) { r[i] = lo[i]; r[i + 8] = hi[i]; }
  return r;
}

// Load a 16-element bf16 fragment slice from two contiguous 16B chunks.
__device__ __forceinline__ v16bf ldfrag(const bf16* p, int off0, int off1) {
  v8bf lo = *(const v8bf*)(p + off0);
  v8bf hi = *(const v8bf*)(p + off1);
  return pack16(lo, hi);
}

// ---------------------------------------------------------------------------
// fp32 -> bf16 convert (grid-stride)
// ---------------------------------------------------------------------------
__global__ void cvt_f32_bf16(const float* __restrict__ in, bf16* __restrict__ out, int n) {
  for (int i = blockIdx.x * blockDim.x + threadIdx.x; i < n; i += gridDim.x * blockDim.x)
    out[i] = (bf16)in[i];
}

// ---------------------------------------------------------------------------
// RoPE (interleaved pairs) fp32 -> bf16.  rows = B*S, cols = heads*HD.
// ---------------------------------------------------------------------------
__global__ void rope_f32_bf16(const float* __restrict__ in, bf16* __restrict__ out,
                              int rows, int cols) {
  const int npairs = rows * (cols >> 1);
  for (int idx = blockIdx.x * blockDim.x + threadIdx.x; idx < npairs;
       idx += gridDim.x * blockDim.x) {
    const int hc  = cols >> 1;
    const int row = idx / hc;
    const int p   = idx - row * hc;
    const int i   = p & 31;        // pair index inside head (HD/2 = 32)
    const int s   = row & 2047;    // position within sequence
    const float inv = __powf(10000.0f, -(2.0f * (float)i) / 64.0f);
    const float ang = (float)s * inv;
    float sn, cs;
    __sincosf(ang, &sn, &cs);
    const float x0 = in[(size_t)row * cols + 2 * p];
    const float x1 = in[(size_t)row * cols + 2 * p + 1];
    out[(size_t)row * cols + 2 * p]     = (bf16)(x0 * cs - x1 * sn);
    out[(size_t)row * cols + 2 * p + 1] = (bf16)(x0 * sn + x1 * cs);
  }
}

// ---------------------------------------------------------------------------
// Tiled bf16 WMMA GEMM:  C[M,N] (fp32) = A[M,K] * B[N,K]^T   (Linear layout).
// Block tile 128x128x32, 8 waves (2x4), each wave owns 64x32 -> 8 wmma/Kstep.
// Double-buffered LDS; tile k+1 copied (async when available) during tile k's
// WMMAs.  Async loads complete in order, so waiting ASYNCcnt<=4 guarantees the
// current tile's 4 copies have landed while next tile's 4 remain in flight.
// ---------------------------------------------------------------------------
__global__ __launch_bounds__(256) void gemm_bf16_wmma(
    const bf16* __restrict__ A, const bf16* __restrict__ B,
    float* __restrict__ C, int M, int N, int K) {
  __shared__ bf16 As[2][128][40];   // 32 used + pad, rows 16B-aligned (80B stride)
  __shared__ bf16 Bs[2][128][40];

  const int t    = threadIdx.x;
  const int wave = t >> 5;
  const int lane = t & 31;
  const int l16  = lane & 15;
  const int hi   = lane >> 4;          // 0: lanes 0-15, 1: lanes 16-31
  const int wm   = wave & 1;           // 2 row-groups of 64
  const int wn   = wave >> 1;          // 4 col-groups of 32
  const int m0   = blockIdx.y * 128;
  const int n0   = blockIdx.x * 128;

  // cooperative tile-copy mapping: thread -> (row, 16-element half)
  const int lr = t >> 1;
  const int lc = (t & 1) * 16;
  const bf16* gA = A + (size_t)(m0 + lr) * K + lc;
  const bf16* gB = B + (size_t)(n0 + lr) * K + lc;

  v8f acc[4][2];
#pragma unroll
  for (int mt = 0; mt < 4; ++mt)
#pragma unroll
    for (int nt = 0; nt < 2; ++nt) acc[mt][nt] = ZERO8;

  const int ac0 = hi ? 16 : 0;   // A-fragment K base (contiguous layout)
  const int bc0 = hi ? 16 : 0;   // B-fragment K base (contiguous layout)

  const int nk = K >> 5;
  // prologue: tile 0 -> buffer 0
  copy16_g2l(gA, &As[0][lr][lc]);
  copy16_g2l(gA + 8, &As[0][lr][lc + 8]);
  copy16_g2l(gB, &Bs[0][lr][lc]);
  copy16_g2l(gB + 8, &Bs[0][lr][lc + 8]);

  for (int kt = 0; kt < nk; ++kt) {
    const int cur = kt & 1;
    if (kt + 1 < nk) {
      const int nx = cur ^ 1;
      const int ko = (kt + 1) << 5;
      copy16_g2l(gA + ko, &As[nx][lr][lc]);
      copy16_g2l(gA + ko + 8, &As[nx][lr][lc + 8]);
      copy16_g2l(gB + ko, &Bs[nx][lr][lc]);
      copy16_g2l(gB + ko + 8, &Bs[nx][lr][lc + 8]);
      wait_async_le4();          // tile kt's 4 copies done; next 4 in flight
    } else {
      wait_async_0();
    }
    __syncthreads();             // all waves' tile-kt data visible

    v16bf af[4], bfr[2];
#pragma unroll
    for (int mt = 0; mt < 4; ++mt) {
      const bf16* rp = &As[cur][wm * 64 + mt * 16 + l16][0];
      af[mt] = ldfrag(rp, ac0, ac0 + 8);
    }
#pragma unroll
    for (int nt = 0; nt < 2; ++nt) {
      const bf16* rp = &Bs[cur][wn * 32 + nt * 16 + l16][0];
      bfr[nt] = ldfrag(rp, bc0, bc0 + 8);
    }
#pragma unroll
    for (int mt = 0; mt < 4; ++mt)
#pragma unroll
      for (int nt = 0; nt < 2; ++nt)
        acc[mt][nt] = __builtin_amdgcn_wmma_f32_16x16x32_bf16(
            false, af[mt], false, bfr[nt], (short)0, acc[mt][nt], false, false);
    __syncthreads();             // done reading buf[cur] before it is re-filled
  }

#pragma unroll
  for (int mt = 0; mt < 4; ++mt)
#pragma unroll
    for (int nt = 0; nt < 2; ++nt)
#pragma unroll
      for (int r = 0; r < 8; ++r) {
        const int row = m0 + wm * 64 + mt * 16 + r + hi * 8;
        const int col = n0 + wn * 32 + nt * 16 + l16;
        C[(size_t)row * N + col] = acc[mt][nt][r];
      }
}

// ---------------------------------------------------------------------------
// Flash attention (full softmax, GQA).  Grid: x = S/128 (query blocks),
// y = B*NH.  8 waves/block, each wave owns 16 query rows; 32-key chunks.
// Q/K already RoPE'd (bf16).  Output written bf16 for the final projection.
// ---------------------------------------------------------------------------
__global__ __launch_bounds__(256) void attn_wmma(
    const bf16* __restrict__ Qb, const bf16* __restrict__ Kb,
    const bf16* __restrict__ Vb, bf16* __restrict__ Ob) {
  const int bh  = blockIdx.y;
  const int b   = bh >> 5;          // NH = 32
  const int h   = bh & 31;
  const int kvh = h >> 2;           // qpk = 4
  const int q0  = blockIdx.x * 128;

  __shared__ bf16 Kc[32][72];       // [key][d]      row stride 144B
  __shared__ bf16 Vt[64][40];       // [d][key]      row stride 80B (transposed)
  __shared__ bf16 Ps[8][16][40];    // per-wave P relayout scratch

  const int t    = threadIdx.x;
  const int wave = t >> 5;
  const int lane = t & 31;
  const int l16  = lane & 15;
  const int hi   = lane >> 4;
  const int ac0  = hi ? 8 : 0;

  // Q fragments for this wave's 16 rows (held in registers for whole kernel).
  const int qrow = b * 2048 + q0 + wave * 16 + l16;
  const bf16* qp = Qb + (size_t)qrow * 2048 + h * 64;
  v16bf qf[2];
#pragma unroll
  for (int kk = 0; kk < 2; ++kk)
    qf[kk] = ldfrag(qp + kk * 32, ac0, ac0 + 16);

  float mrun[8], lrun[8];
  v8f o[4];
#pragma unroll
  for (int r = 0; r < 8; ++r) { mrun[r] = -1e30f; lrun[r] = 0.0f; }
#pragma unroll
  for (int t4 = 0; t4 < 4; ++t4) o[t4] = ZERO8;

  const float scale = 0.125f;       // 1/sqrt(64)
  const int srow = t >> 3;          // staging: 0..31 key rows
  const int sch  = t & 7;           // staging: 8-elem chunk of HD

  for (int key0 = 0; key0 < 2048; key0 += 32) {
    // ---- stage K chunk [32x64] (async) and V transposed [64x32] into LDS ----
    {
      const size_t base = (size_t)(b * 2048 + key0 + srow) * 512 + kvh * 64 + sch * 8;
      copy16_g2l(Kb + base, &Kc[srow][sch * 8]);
      uint4 vq = *(const uint4*)(Vb + base);
      const bf16* ve = (const bf16*)&vq;
#pragma unroll
      for (int j = 0; j < 8; ++j) Vt[sch * 8 + j][srow] = ve[j];
    }
    wait_async_0();
    __syncthreads();

    // ---- scores: S[16q x 32k] = Q(16x64) @ Kchunk(32x64)^T ----
    v8f s0 = ZERO8, s1 = ZERO8;
#pragma unroll
    for (int kk = 0; kk < 2; ++kk) {
      v16bf b0 = ldfrag(&Kc[l16][kk * 32], ac0, ac0 + 16);       // keys 0-15
      v16bf b1 = ldfrag(&Kc[16 + l16][kk * 32], ac0, ac0 + 16);  // keys 16-31
      s0 = __builtin_amdgcn_wmma_f32_16x16x32_bf16(false, qf[kk], false, b0,
                                                   (short)0, s0, false, false);
      s1 = __builtin_amdgcn_wmma_f32_16x16x32_bf16(false, qf[kk], false, b1,
                                                   (short)0, s1, false, false);
    }

    // ---- online softmax (per-row reduction across 16-lane halves) ----
    float pr0[8], pr1[8], corr[8];
#pragma unroll
    for (int r = 0; r < 8; ++r) {
      float a0 = s0[r] * scale, a1 = s1[r] * scale;
      float mx = fmaxf(a0, a1);
#pragma unroll
      for (int m = 1; m < 16; m <<= 1) mx = fmaxf(mx, __shfl_xor(mx, m, 16));
      const float mnew = fmaxf(mrun[r], mx);
      corr[r] = __expf(mrun[r] - mnew);
      mrun[r] = mnew;
      const float p0 = __expf(a0 - mnew);
      const float p1 = __expf(a1 - mnew);
      float rs = p0 + p1;
#pragma unroll
      for (int m = 1; m < 16; m <<= 1) rs += __shfl_xor(rs, m, 16);
      lrun[r] = lrun[r] * corr[r] + rs;
      pr0[r] = p0; pr1[r] = p1;
    }
#pragma unroll
    for (int t4 = 0; t4 < 4; ++t4)
#pragma unroll
      for (int r = 0; r < 8; ++r) o[t4][r] *= corr[r];

    // ---- relayout P (C-frag -> A-frag) through per-wave LDS scratch ----
#pragma unroll
    for (int r = 0; r < 8; ++r) {
      const int m = r + hi * 8;
      Ps[wave][m][l16]      = (bf16)pr0[r];
      Ps[wave][m][16 + l16] = (bf16)pr1[r];
    }
    asm volatile("s_wait_dscnt 0" ::: "memory");  // DS in-order per wave; wait data
    v16bf pf = ldfrag(&Ps[wave][l16][0], ac0, ac0 + 16);

    // ---- O += P(16x32) @ Vchunk(32x64), 4 N-tiles ----
#pragma unroll
    for (int t4 = 0; t4 < 4; ++t4) {
      v16bf vf = ldfrag(&Vt[t4 * 16 + l16][0], ac0, ac0 + 16);
      o[t4] = __builtin_amdgcn_wmma_f32_16x16x32_bf16(false, pf, false, vf,
                                                      (short)0, o[t4], false, false);
    }
    __syncthreads();   // protect Kc/Vt before next staging
  }

  // ---- epilogue: normalize, store bf16 [B*S, NH*HD] ----
#pragma unroll
  for (int t4 = 0; t4 < 4; ++t4)
#pragma unroll
    for (int r = 0; r < 8; ++r) {
      const int m  = r + hi * 8;
      const int sr = q0 + wave * 16 + m;
      Ob[(size_t)(b * 2048 + sr) * 2048 + h * 64 + t4 * 16 + l16] =
          (bf16)(o[t4][r] / lrun[r]);
    }
}

// ---------------------------------------------------------------------------
// Host-side launch
// ---------------------------------------------------------------------------
extern "C" void kernel_launch(void* const* d_in, const int* in_sizes, int n_in,
                              void* d_out, int out_size, void* d_ws, size_t ws_size,
                              hipStream_t stream) {
  (void)in_sizes; (void)n_in; (void)out_size; (void)ws_size;
  const float* x  = (const float*)d_in[0];
  const float* Wq = (const float*)d_in[1];
  const float* Wk = (const float*)d_in[2];
  const float* Wv = (const float*)d_in[3];
  const float* Wo = (const float*)d_in[4];
  float* out = (float*)d_out;

  const int BS = 2 * 2048;          // B*S = 4096
  const int D  = 2048;
  const int DK = 512;               // NKV*HD

  // workspace carve-up (256B aligned)
  char* ws = (char*)d_ws;
  size_t off = 0;
  auto carve = [&](size_t bytes) -> char* {
    char* p = ws + off;
    off += (bytes + 255) & ~(size_t)255;
    return p;
  };
  bf16*  xb  = (bf16*)carve((size_t)BS * D * 2);
  bf16*  Wqb = (bf16*)carve((size_t)D * D * 2);
  bf16*  Wkb = (bf16*)carve((size_t)DK * D * 2);
  bf16*  Wvb = (bf16*)carve((size_t)DK * D * 2);
  bf16*  Wob = (bf16*)carve((size_t)D * D * 2);
  float* Qf  = (float*)carve((size_t)BS * D * 4);
  float* Kf  = (float*)carve((size_t)BS * DK * 4);
  float* Vf  = (float*)carve((size_t)BS * DK * 4);
  bf16*  Qbb = (bf16*)carve((size_t)BS * D * 2);
  bf16*  Kbb = (bf16*)carve((size_t)BS * DK * 2);
  bf16*  Vbb = (bf16*)carve((size_t)BS * DK * 2);
  bf16*  AOb = (bf16*)carve((size_t)BS * D * 2);

  const int TB = 256;
  auto cgrid = [](int n) { int g = (n + 255) / 256; return g > 8192 ? 8192 : g; };

  // 1) fp32 -> bf16 conversions
  cvt_f32_bf16<<<cgrid(BS * D), TB, 0, stream>>>(x,  xb,  BS * D);
  cvt_f32_bf16<<<cgrid(D * D),  TB, 0, stream>>>(Wq, Wqb, D * D);
  cvt_f32_bf16<<<cgrid(DK * D), TB, 0, stream>>>(Wk, Wkb, DK * D);
  cvt_f32_bf16<<<cgrid(DK * D), TB, 0, stream>>>(Wv, Wvb, DK * D);
  cvt_f32_bf16<<<cgrid(D * D),  TB, 0, stream>>>(Wo, Wob, D * D);

  // 2) projections (WMMA GEMMs)
  gemm_bf16_wmma<<<dim3(D / 128,  BS / 128), TB, 0, stream>>>(xb, Wqb, Qf, BS, D,  D);
  gemm_bf16_wmma<<<dim3(DK / 128, BS / 128), TB, 0, stream>>>(xb, Wkb, Kf, BS, DK, D);
  gemm_bf16_wmma<<<dim3(DK / 128, BS / 128), TB, 0, stream>>>(xb, Wvb, Vf, BS, DK, D);

  // 3) RoPE on Q and K (-> bf16), V straight convert
  rope_f32_bf16<<<cgrid(BS * D / 2),  TB, 0, stream>>>(Qf, Qbb, BS, D);
  rope_f32_bf16<<<cgrid(BS * DK / 2), TB, 0, stream>>>(Kf, Kbb, BS, DK);
  cvt_f32_bf16<<<cgrid(BS * DK),      TB, 0, stream>>>(Vf, Vbb, BS * DK);

  // 4) flash attention (grid: 16 query blocks x 64 (b,head) pairs)
  attn_wmma<<<dim3(2048 / 128, 2 * 32), TB, 0, stream>>>(Qbb, Kbb, Vbb, AOb);

  // 5) output projection -> fp32 d_out
  gemm_bf16_wmma<<<dim3(D / 128, BS / 128), TB, 0, stream>>>(AOb, Wob, out, BS, D, D);
}
